// SCSA_36301063586002
// MI455X (gfx1250) — matrix operand order repaired and checked by reading, not compiled, not run on hardware
//
#include <hip/hip_runtime.h>

// ---------------- problem dims ----------------
#define BB 2
#define TT 8
#define NB (BB*TT)      // 16   (B*T)
#define CC 256
#define NH 4            // heads
#define SS 4096         // H*W
#define C3 768
#define KKDIM 768       // 3 taps * 256 channels

typedef __attribute__((ext_vector_type(16))) __bf16 v16bf;
typedef __attribute__((ext_vector_type(8)))  float  v8f;

union U16B { uint4 q; __bf16 h[8]; };

// Async global->LDS helpers (CDNA5): one b128 per lane, tracked by ASYNCcnt.
__device__ __forceinline__ void async_g2l_b128(const void* gptr, void* lptr) {
  unsigned la = (unsigned)(size_t)lptr;                 // low 32 bits of LDS aperture addr == LDS offset
  unsigned long long ga = (unsigned long long)(size_t)gptr;
  asm volatile("global_load_async_to_lds_b128 %0, %1, off" :: "v"(la), "v"(ga) : "memory");
}
__device__ __forceinline__ void wait_async0() { asm volatile("s_wait_asynccnt 0x0" ::: "memory"); }
__device__ __forceinline__ void wait_async1() { asm volatile("s_wait_asynccnt 0x1" ::: "memory"); }

// ---------------- workspace layout (bytes) ----------------
static constexpr size_t OFF_XBF  = 0;                               // reused as attention-output buffer
static constexpr size_t SZ_XBF   = (size_t)NB*CC*SS*2;              // 32 MiB
static constexpr size_t OFF_WIN  = OFF_XBF + SZ_XBF;
static constexpr size_t SZ_WIN   = (size_t)C3*KKDIM*2;
static constexpr size_t OFF_WOUT = OFF_WIN + SZ_WIN;
static constexpr size_t SZ_WOUT  = (size_t)CC*KKDIM*2;
static constexpr size_t OFF_RINV = OFF_WOUT + SZ_WOUT;
static constexpr size_t SZ_RINV  = (size_t)NB*512*4;
static constexpr size_t OFF_ATTN = OFF_RINV + SZ_RINV;
static constexpr size_t SZ_ATTN  = (size_t)NB*NH*64*64*2;
static constexpr size_t OFF_YBF  = OFF_ATTN + SZ_ATTN;
static constexpr size_t SZ_YBF   = (size_t)NB*C3*SS*2;              // 96 MiB
static constexpr size_t OFF_QKV  = OFF_YBF + SZ_YBF;                // 96 MiB

// ---------------- K0: fp32 -> bf16 convert of x ----------------
__global__ void k_cvt_x(const float* __restrict__ x, __bf16* __restrict__ xbf) {
  size_t i = ((size_t)blockIdx.x * blockDim.x + threadIdx.x) * 4;
  float4 v = *(const float4*)(x + i);
  union { __bf16 h[4]; uint2 u; } o;
  o.h[0] = (__bf16)v.x; o.h[1] = (__bf16)v.y; o.h[2] = (__bf16)v.z; o.h[3] = (__bf16)v.w;
  *(uint2*)(xbf + i) = o.u;
}

// ---------------- K0b: pack temporal-conv weights [M][K], K = dt*256+ic ----------------
__global__ void k_pack_w(const float* __restrict__ w, __bf16* __restrict__ out, int M) {
  int idx = blockIdx.x * 256 + threadIdx.x;
  if (idx >= M * KKDIM) return;
  int m = idx / KKDIM, k = idx % KKDIM;
  int dt = k >> 8, ic = k & 255;
  out[idx] = (__bf16)w[(m * CC + ic) * 3 + dt];
}

// ---------------- K1/K5: temporal conv as WMMA GEMM, async double-buffered B tile ----------------
// out[n][m][s] = bias[m] + sum_k Wm[m][k] * src[b*T + (t + k/256 - 1)][k%256][s]   (zero outside [0,T))
template<bool OUTF32>
__global__ __launch_bounds__(256) void k_tconv_gemm(const __bf16* __restrict__ Wm,
                                                    const float*  __restrict__ bias,
                                                    const __bf16* __restrict__ src,
                                                    void* __restrict__ dstv, int Mtot) {
  const int n = blockIdx.z, b = n / TT, t = n % TT;
  const int s0 = blockIdx.x * 64;
  const int w = threadIdx.x >> 5, lane = threadIdx.x & 31, half = lane >> 4, ln = lane & 15;
  const int m0 = blockIdx.y * 128 + w * 16;

  __shared__ __align__(16) __bf16 Bs[2][32 * 64];

  v8f acc[4];
  for (int i = 0; i < 4; ++i) for (int j = 0; j < 8; ++j) acc[i][j] = 0.f;

  const int ldr = threadIdx.x >> 3;        // 0..31 : K-row within tile
  const int ldc = (threadIdx.x & 7) * 8;   // column chunk

  // issue tile ks into buffer bi; returns true if an async load was issued (per wave: 1 b128/lane)
  auto issue = [&](int ks, int bi) -> bool {
    const int k0 = ks * 32;
    const int dt = k0 >> 8;                // constant over the 32-row span
    const int ic0 = k0 & 255;
    const int ts = t + dt - 1;
    __bf16* dst = &Bs[bi][ldr * 64 + ldc];
    if (ts >= 0 && ts < TT) {
      const __bf16* gsrc = src + ((size_t)(b * TT + ts) * CC + ic0 + ldr) * SS + s0 + ldc;
      async_g2l_b128(gsrc, dst);
      return true;
    }
    uint4 z; z.x = z.y = z.z = z.w = 0u;
    *(uint4*)dst = z;                      // zero-pad tile (t boundary)
    return false;
  };

  constexpr int NSTEP = KKDIM / 32;        // 24
  issue(0, 0);
  for (int ks = 0; ks < NSTEP; ++ks) {
    const int cur = ks & 1;
    const int k0 = ks * 32;
    bool nxt_async = false;
    if (ks + 1 < NSTEP) nxt_async = issue(ks + 1, cur ^ 1);
    // async loads complete in order: allow the next tile's load to stay in flight
    if (nxt_async) wait_async1(); else wait_async0();
    __syncthreads();

    v16bf a;
    {
      const __bf16* ap = Wm + (size_t)(m0 + ln) * KKDIM + k0 + half * 16;
      U16B u0, u1; u0.q = *(const uint4*)ap; u1.q = *(const uint4*)(ap + 8);
      for (int j = 0; j < 8; ++j) { a[j] = u0.h[j]; a[8 + j] = u1.h[j]; }
      if (k0 + 32 < KKDIM) __builtin_prefetch(ap + 32, 0, 0);   // global_prefetch next A chunk
    }
    for (int sub = 0; sub < 4; ++sub) {
      v16bf bf;
      for (int j = 0; j < 16; ++j) bf[j] = Bs[cur][(half * 16 + j) * 64 + sub * 16 + ln];
      acc[sub] = __builtin_amdgcn_wmma_f32_16x16x32_bf16(false, a, false, bf, (short)0,
                                                         acc[sub], false, false);
    }
    __syncthreads();
  }

  for (int sub = 0; sub < 4; ++sub)
    for (int j = 0; j < 8; ++j) {
      const int row = m0 + half * 8 + j;
      const int col = s0 + sub * 16 + ln;
      const float v = acc[sub][j] + bias[row];
      if (OUTF32) ((float*)dstv)[((size_t)n * Mtot + row) * SS + col] = v;
      else        ((__bf16*)dstv)[((size_t)n * Mtot + row) * SS + col] = (__bf16)v;
    }
}

// ---------------- K2: depthwise 3x3 + row sum-of-squares for q,k ----------------
__global__ __launch_bounds__(256) void k_dwconv(const __bf16* __restrict__ ybf,
                                                const float* __restrict__ dw,
                                                const float* __restrict__ db,
                                                __bf16* __restrict__ qkv,
                                                float* __restrict__ rinv) {
  const int n = blockIdx.x / C3;
  const int ch = blockIdx.x % C3;
  const __bf16* plane = ybf + ((size_t)n * C3 + ch) * SS;
  __shared__ __align__(16) __bf16 tile[64 * 64];
  __shared__ float red[256];
  {
    const int off = threadIdx.x * 16;
    async_g2l_b128(plane + off,     &tile[off]);
    async_g2l_b128(plane + off + 8, &tile[off + 8]);
    wait_async0();
  }
  __syncthreads();
  float wv[9];
  for (int i = 0; i < 9; ++i) wv[i] = dw[ch * 9 + i];
  const float bias = db[ch];
  __bf16* orow = qkv + ((size_t)n * C3 + ch) * SS;
  float ss = 0.f;
  for (int i = 0; i < 16; ++i) {
    const int p = threadIdx.x * 16 + i;
    const int y = p >> 6, x = p & 63;
    float r = bias;
    for (int dy = 0; dy < 3; ++dy) {
      const int yy = y + dy - 1; if (yy < 0 || yy > 63) continue;
      for (int dx = 0; dx < 3; ++dx) {
        const int xx = x + dx - 1; if (xx < 0 || xx > 63) continue;
        r += (float)tile[yy * 64 + xx] * wv[dy * 3 + dx];
      }
    }
    orow[p] = (__bf16)r;
    ss += r * r;
  }
  red[threadIdx.x] = ss;
  __syncthreads();
  for (int st = 128; st > 0; st >>= 1) {
    if ((int)threadIdx.x < st) red[threadIdx.x] += red[threadIdx.x + st];
    __syncthreads();
  }
  if (threadIdx.x == 0 && ch < 512)
    rinv[n * 512 + ch] = 1.f / fmaxf(sqrtf(red[0]), 1e-12f);
}

// ---------------- K3: attn = softmax(beta * qn.knT)*w0 + relu(.)^2*w1 ----------------
__global__ __launch_bounds__(256) void k_attn(const __bf16* __restrict__ qkv,
                                              const float* __restrict__ rinv,
                                              const float* __restrict__ beta,
                                              const float* __restrict__ wmix,
                                              __bf16* __restrict__ attn_bf) {
  const int nh = blockIdx.x, n = nh >> 2, hd = nh & 3;
  const int w = threadIdx.x >> 5, lane = threadIdx.x & 31, half = lane >> 4, ln = lane & 15;
  const int mi = w & 3, nb0 = (w >> 2) * 2;

  __shared__ float At[64 * 64];

  const int cr = mi * 16 + ln;
  const float sq = rinv[n * 512 + cr * NH + hd];
  const int d0 = nb0 * 16 + ln, d1 = d0 + 16;
  const float sk0 = rinv[n * 512 + 256 + d0 * NH + hd];
  const float sk1 = rinv[n * 512 + 256 + d1 * NH + hd];
  const __bf16* qrow  = qkv + ((size_t)n * C3 + cr * NH + hd) * SS;
  const __bf16* krow0 = qkv + ((size_t)n * C3 + 256 + d0 * NH + hd) * SS;
  const __bf16* krow1 = qkv + ((size_t)n * C3 + 256 + d1 * NH + hd) * SS;

  v8f acc0, acc1;
  for (int j = 0; j < 8; ++j) { acc0[j] = 0.f; acc1[j] = 0.f; }

  for (int k0 = 0; k0 < SS; k0 += 32) {
    U16B a0, a1;
    a0.q = *(const uint4*)(qrow + k0 + half * 16);
    a1.q = *(const uint4*)(qrow + k0 + half * 16 + 8);
    v16bf a;
    for (int j = 0; j < 8; ++j) {
      a[j]     = (__bf16)((float)a0.h[j] * sq);
      a[8 + j] = (__bf16)((float)a1.h[j] * sq);
    }
    U16B b00, b01, b10, b11;
    b00.q = *(const uint4*)(krow0 + k0 + half * 16);
    b01.q = *(const uint4*)(krow0 + k0 + half * 16 + 8);
    b10.q = *(const uint4*)(krow1 + k0 + half * 16);
    b11.q = *(const uint4*)(krow1 + k0 + half * 16 + 8);
    v16bf b0, b1;
    for (int j = 0; j < 8; ++j) {
      b0[j] = (__bf16)((float)b00.h[j] * sk0); b0[8 + j] = (__bf16)((float)b01.h[j] * sk0);
      b1[j] = (__bf16)((float)b10.h[j] * sk1); b1[8 + j] = (__bf16)((float)b11.h[j] * sk1);
    }
    acc0 = __builtin_amdgcn_wmma_f32_16x16x32_bf16(false, a, false, b0, (short)0, acc0, false, false);
    acc1 = __builtin_amdgcn_wmma_f32_16x16x32_bf16(false, a, false, b1, (short)0, acc1, false, false);
  }

  const float bet = beta[hd];
  for (int j = 0; j < 8; ++j) {
    const int row = mi * 16 + half * 8 + j;
    At[row * 64 + nb0 * 16 + ln]       = acc0[j] * bet;
    At[row * 64 + (nb0 + 1) * 16 + ln] = acc1[j] * bet;
  }
  __syncthreads();

  if (threadIdx.x < 64) {
    const int r = threadIdx.x;
    const float ww0 = wmix[0], ww1 = wmix[1];
    const float wm = fmaxf(ww0, ww1);
    const float e0 = expf(ww0 - wm), e1 = expf(ww1 - wm);
    const float w0 = e0 / (e0 + e1), w1 = e1 / (e0 + e1);
    float mx = -3.4e38f;
    for (int d = 0; d < 64; ++d) mx = fmaxf(mx, At[r * 64 + d]);
    float sum = 0.f;
    for (int d = 0; d < 64; ++d) sum += expf(At[r * 64 + d] - mx);
    const float isum = 1.f / sum;
    for (int d = 0; d < 64; ++d) {
      const float v = At[r * 64 + d];
      const float sm = expf(v - mx) * isum;
      const float rl = fmaxf(v, 0.f);
      attn_bf[(size_t)nh * 4096 + r * 64 + d] = (__bf16)(w0 * sm + w1 * rl * rl);
    }
  }
}

// ---------------- K4: out = attn @ v ;  output channel = hd*64 + cr (reference quirk) ----------------
__global__ __launch_bounds__(256) void k_attn_out(const __bf16* __restrict__ attn_bf,
                                                  const __bf16* __restrict__ qkv,
                                                  __bf16* __restrict__ aout) {
  const int nh = blockIdx.y, n = nh >> 2, hd = nh & 3;
  const int s0 = blockIdx.x * 64;
  const int w = threadIdx.x >> 5, lane = threadIdx.x & 31, half = lane >> 4, ln = lane & 15;
  const int mi = w & 3, nb0 = (w >> 2) * 2;

  __shared__ __align__(16) __bf16 Vs[64 * 64];
  {
    const int d = threadIdx.x >> 2;
    const int co = (threadIdx.x & 3) * 16;
    const __bf16* vp = qkv + ((size_t)n * C3 + 512 + d * NH + hd) * SS + s0 + co;
    async_g2l_b128(vp,     &Vs[d * 64 + co]);
    async_g2l_b128(vp + 8, &Vs[d * 64 + co + 8]);
    wait_async0();
  }
  __syncthreads();

  v8f acc0, acc1;
  for (int j = 0; j < 8; ++j) { acc0[j] = 0.f; acc1[j] = 0.f; }

  const __bf16* arow = attn_bf + (size_t)nh * 4096 + (mi * 16 + ln) * 64;
  for (int k0 = 0; k0 < 64; k0 += 32) {
    U16B u0, u1;
    u0.q = *(const uint4*)(arow + k0 + half * 16);
    u1.q = *(const uint4*)(arow + k0 + half * 16 + 8);
    v16bf a;
    for (int j = 0; j < 8; ++j) { a[j] = u0.h[j]; a[8 + j] = u1.h[j]; }
    v16bf b0, b1;
    for (int j = 0; j < 16; ++j) {
      b0[j] = Vs[(k0 + half * 16 + j) * 64 + nb0 * 16 + ln];
      b1[j] = Vs[(k0 + half * 16 + j) * 64 + (nb0 + 1) * 16 + ln];
    }
    acc0 = __builtin_amdgcn_wmma_f32_16x16x32_bf16(false, a, false, b0, (short)0, acc0, false, false);
    acc1 = __builtin_amdgcn_wmma_f32_16x16x32_bf16(false, a, false, b1, (short)0, acc1, false, false);
  }

  for (int j = 0; j < 8; ++j) {
    const int row = mi * 16 + half * 8 + j;
    aout[((size_t)n * CC + hd * 64 + row) * SS + s0 + nb0 * 16 + ln]       = (__bf16)acc0[j];
    aout[((size_t)n * CC + hd * 64 + row) * SS + s0 + (nb0 + 1) * 16 + ln] = (__bf16)acc1[j];
  }
}

// ---------------- launcher ----------------
extern "C" void kernel_launch(void* const* d_in, const int* in_sizes, int n_in,
                              void* d_out, int out_size, void* d_ws, size_t ws_size,
                              hipStream_t stream) {
  const float* x          = (const float*)d_in[0];
  const float* conv_in_w  = (const float*)d_in[1];
  const float* conv_in_b  = (const float*)d_in[2];
  const float* dconv_w    = (const float*)d_in[3];
  const float* dconv_b    = (const float*)d_in[4];
  const float* conv_out_w = (const float*)d_in[5];
  const float* conv_out_b = (const float*)d_in[6];
  const float* beta       = (const float*)d_in[7];
  const float* weight     = (const float*)d_in[8];

  char* ws = (char*)d_ws;
  __bf16* xbf  = (__bf16*)(ws + OFF_XBF);
  __bf16* win  = (__bf16*)(ws + OFF_WIN);
  __bf16* wout = (__bf16*)(ws + OFF_WOUT);
  float*  rinv = (float*)(ws + OFF_RINV);
  __bf16* attn = (__bf16*)(ws + OFF_ATTN);
  __bf16* ybf  = (__bf16*)(ws + OFF_YBF);
  __bf16* qkv  = (__bf16*)(ws + OFF_QKV);
  __bf16* aout = xbf;  // xbf dead after K1; reuse for attention output

  const size_t xelem = (size_t)NB * CC * SS;  // 16.7M
  k_cvt_x<<<(unsigned)(xelem / 4 / 256), 256, 0, stream>>>(x, xbf);
  k_pack_w<<<(C3 * KKDIM + 255) / 256, 256, 0, stream>>>(conv_in_w, win, C3);
  k_pack_w<<<(CC * KKDIM + 255) / 256, 256, 0, stream>>>(conv_out_w, wout, CC);

  k_tconv_gemm<false><<<dim3(SS / 64, C3 / 128, NB), 256, 0, stream>>>(win, conv_in_b, xbf, (void*)ybf, C3);
  k_dwconv<<<NB * C3, 256, 0, stream>>>(ybf, dconv_w, dconv_b, qkv, rinv);
  k_attn<<<NB * NH, 256, 0, stream>>>(qkv, rinv, beta, weight, attn);
  k_attn_out<<<dim3(SS / 64, NB * NH), 256, 0, stream>>>(attn, qkv, aout);
  k_tconv_gemm<true><<<dim3(SS / 64, CC / 128, NB), 256, 0, stream>>>(wout, conv_out_b, aout, d_out, CC);
}
